// PerceiverAttention_47124381172299
// MI455X (gfx1250) — compile-verified, matrix operand order checked by reading, not run on hardware
//
#include <hip/hip_runtime.h>

// ---------------- problem constants ----------------
#define BATCH   4
#define TSEQ    4096
#define NQ      512
#define DIMC    1024
#define HEADS   16
#define DH      64
#define INNER   1024
#define STOT    (TSEQ + NQ)        // 4608

typedef _Float16 h8   __attribute__((ext_vector_type(8)));
typedef _Float16 v16h __attribute__((ext_vector_type(16)));
typedef float    v8f  __attribute__((ext_vector_type(8)));

// Build a 16xK(32) WMMA fragment per the CDNA5 16-bit A/B layout:
// lane holds 8 halves at k = half*8.. and 8 at k = 16+half*8..  -> two b128 loads
__device__ __forceinline__ v16h ld_frag(const _Float16* p) {
    h8 lo = *(const h8*)(p);
    h8 hi = *(const h8*)(p + 16);
    v16h r;
#pragma unroll
    for (int i = 0; i < 8; ++i) { r[i] = lo[i]; r[i + 8] = hi[i]; }
    return r;
}

__device__ __forceinline__ v8f wmma_f16(v16h a, v16h b, v8f c) {
    return __builtin_amdgcn_wmma_f32_16x16x32_f16(
        /*neg_a=*/false, a, /*neg_b=*/false, b,
        /*c_mod=*/(short)0, c, /*reuse_a=*/false, /*reuse_b=*/false);
}

// ---------------- transpose + convert: W[K][N] f32 -> Wt[N][K] f16 --------------
// grid: (K/64, N/64); 256 threads; tiled via LDS so both sides are vectorized.
__global__ __launch_bounds__(256) void transpose_convert_kernel(
    const float* __restrict__ src, _Float16* __restrict__ dst,
    int K, int N, float scale) {
    __shared__ alignas(16) _Float16 t[64][72];
    const int tid = threadIdx.x;
    const int k0 = blockIdx.x * 64, n0 = blockIdx.y * 64;
    {
        const int kl = tid >> 2, nseg = (tid & 3) * 16;
        const float* sp = src + (long)(k0 + kl) * N + n0 + nseg;
#pragma unroll
        for (int e = 0; e < 16; ++e) t[nseg + e][kl] = (_Float16)(sp[e] * scale);
    }
    __syncthreads();
    {
        const int nl = tid >> 2, kseg = (tid & 3) * 16;
        _Float16* dp = dst + (long)(n0 + nl) * K + k0 + kseg;
        *(h8*)dp       = *(const h8*)&t[nl][kseg];
        *(h8*)(dp + 8) = *(const h8*)&t[nl][kseg + 8];
    }
}

// ---------------- V transpose: kvbuf v-half -> Vt[b][h][DH][STOT] ---------------
// grid: (STOT/64, HEADS, BATCH); 256 threads.
__global__ __launch_bounds__(256) void transpose_v_kernel(
    const _Float16* __restrict__ kvbuf, _Float16* __restrict__ vt) {
    __shared__ alignas(16) _Float16 t[64][72];
    const int tid = threadIdx.x;
    const int s0 = blockIdx.x * 64, h = blockIdx.y, b = blockIdx.z;
    {
        const int sl = tid >> 2, dseg = (tid & 3) * 16;
        const _Float16* kp =
            kvbuf + ((long)(b * STOT + s0 + sl)) * (2 * INNER) + INNER + h * DH + dseg;
        h8 v0 = *(const h8*)kp;
        h8 v1 = *(const h8*)(kp + 8);
#pragma unroll
        for (int e = 0; e < 8; ++e) {
            t[dseg + e][sl]     = v0[e];
            t[dseg + 8 + e][sl] = v1[e];
        }
    }
    __syncthreads();
    {
        const int dl = tid >> 2, sseg = (tid & 3) * 16;
        _Float16* dp = vt + ((long)(b * HEADS + h) * DH + dl) * STOT + s0 + sseg;
        *(h8*)dp       = *(const h8*)&t[dl][sseg];
        *(h8*)(dp + 8) = *(const h8*)&t[dl][sseg + 8];
    }
}

// ---------------- fused LayerNorm for x and queries -> kv_input (f16) ----------
__global__ __launch_bounds__(256) void layernorm_kernel(
    const float* __restrict__ x, const float* __restrict__ q,
    const float* __restrict__ gx, const float* __restrict__ bx,
    const float* __restrict__ gq, const float* __restrict__ bq,
    _Float16* __restrict__ kvin) {
    const int row = blockIdx.x;               // 0 .. BATCH*STOT-1
    const int b   = row / STOT;
    const int s   = row % STOT;
    const float* src; const float* g; const float* be;
    if (s < TSEQ) { src = x + ((long)b * TSEQ + s) * DIMC; g = gx; be = bx; }
    else          { src = q + ((long)b * NQ + (s - TSEQ)) * DIMC; g = gq; be = bq; }

    float v[4]; float sum = 0.f, sq = 0.f;
#pragma unroll
    for (int i = 0; i < 4; ++i) {
        v[i] = src[threadIdx.x + i * 256];
        sum += v[i]; sq += v[i] * v[i];
    }
#pragma unroll
    for (int m = 16; m >= 1; m >>= 1) {
        sum += __shfl_xor(sum, m, 32);
        sq  += __shfl_xor(sq,  m, 32);
    }
    __shared__ float rsum[8], rsq[8];
    const int wave = threadIdx.x >> 5;
    if ((threadIdx.x & 31) == 0) { rsum[wave] = sum; rsq[wave] = sq; }
    __syncthreads();
    float ts = 0.f, tq = 0.f;
#pragma unroll
    for (int w = 0; w < 8; ++w) { ts += rsum[w]; tq += rsq[w]; }
    const float mean = ts * (1.0f / DIMC);
    const float var  = tq * (1.0f / DIMC) - mean * mean;
    const float inv  = rsqrtf(var + 1e-5f);
    _Float16* out = kvin + (long)row * DIMC;
#pragma unroll
    for (int i = 0; i < 4; ++i) {
        int c = threadIdx.x + i * 256;
        out[c] = (_Float16)((v[i] - mean) * inv * g[c] + be[c]);
    }
}

// ---------------- batched f16 GEMM: C[M,N] = A[M,K] * Bt[N][K]^T ----------------
// 128x128 block tile, K-step 64 (2 WMMA slices), 8 waves (2 M x 4 N), 64x32/wave.
// Double-buffered LDS; both tiles stage with b128 stores (B is pre-transposed).
template <bool F16OUT>
__global__ __launch_bounds__(256) void gemm_f16_kernel(
    const _Float16* __restrict__ A, long aBStride, int lda,
    const _Float16* __restrict__ Bt, int ldb,
    void* __restrict__ Cp, int ldc, long cBStride, int K) {
    __shared__ alignas(16) _Float16 sA[2][128][72];   // [m][k], padded
    __shared__ alignas(16) _Float16 sB[2][128][72];   // [n][k], padded

    const int tid  = threadIdx.x;
    const int lane = tid & 31;
    const int wave = tid >> 5;
    const int wm   = wave & 1;          // 0..1  (M dir)
    const int wn   = wave >> 1;         // 0..3  (N dir)
    const int half = lane >> 4;
    const int l15  = lane & 15;
    const int z    = blockIdx.z;

    // global load indices: each thread owns 32 contiguous f16 of one row
    const int row = tid >> 1;                  // 0..127
    const int seg = (tid & 1) * 32;            // 0 / 32

    const _Float16* ap =
        A + (long)z * aBStride + (long)(blockIdx.y * 128 + row) * lda + seg;
    const _Float16* bp = Bt + (long)(blockIdx.x * 128 + row) * ldb + seg;

    auto stage = [&](_Float16* dst, const _Float16* gsrc) {
#pragma unroll
        for (int u = 0; u < 4; ++u)
            *(h8*)(dst + u * 8) = *(const h8*)(gsrc + u * 8);
    };

    v8f acc[4][2];
#pragma unroll
    for (int mt = 0; mt < 4; ++mt)
#pragma unroll
        for (int nt = 0; nt < 2; ++nt) acc[mt][nt] = {};

    // ---- prologue: tile 0 into buffer 0
    stage(&sA[0][row][seg], ap);
    stage(&sB[0][row][seg], bp);
    __syncthreads();

    const int nIter = K >> 6;
    int buf = 0;
    for (int kt = 0; kt < nIter; ++kt) {
        const bool hasNext = (kt + 1 < nIter);
        h8 na[4], nb[4];
        if (hasNext) {
            const _Float16* ap2 = ap + (kt + 1) * 64;
            const _Float16* bp2 = bp + (kt + 1) * 64;
#pragma unroll
            for (int u = 0; u < 4; ++u) {
                na[u] = *(const h8*)(ap2 + u * 8);
                nb[u] = *(const h8*)(bp2 + u * 8);
            }
            if (kt + 2 < nIter) {
                __builtin_prefetch(ap + (kt + 2) * 64, 0, 1);
                __builtin_prefetch(bp + (kt + 2) * 64, 0, 1);
            }
        }

#pragma unroll
        for (int kk = 0; kk < 2; ++kk) {
            v16h aF[4], bF[2];
#pragma unroll
            for (int mt = 0; mt < 4; ++mt) {
                int rowm = wm * 64 + mt * 16 + l15;
                aF[mt] = ld_frag(&sA[buf][rowm][kk * 32 + half * 8]);
            }
#pragma unroll
            for (int nt = 0; nt < 2; ++nt) {
                int coln = wn * 32 + nt * 16 + l15;
                bF[nt] = ld_frag(&sB[buf][coln][kk * 32 + half * 8]);
            }
#pragma unroll
            for (int mt = 0; mt < 4; ++mt)
#pragma unroll
                for (int nt = 0; nt < 2; ++nt)
                    acc[mt][nt] = wmma_f16(aF[mt], bF[nt], acc[mt][nt]);
        }

        if (hasNext) {
            _Float16* da = &sA[buf ^ 1][row][seg];
            _Float16* db = &sB[buf ^ 1][row][seg];
#pragma unroll
            for (int u = 0; u < 4; ++u) {
                *(h8*)(da + u * 8) = na[u];
                *(h8*)(db + u * 8) = nb[u];
            }
            __syncthreads();
            buf ^= 1;
        }
    }

    // ---- branch-free epilogue (C layout: lanes 0-15 m=r, lanes 16-31 m=8+r, n=l15)
    const int mBase = blockIdx.y * 128 + wm * 64 + half * 8;
    const int nBase = blockIdx.x * 128 + wn * 32 + l15;
    if constexpr (F16OUT) {
        _Float16* C = (_Float16*)Cp + (long)z * cBStride;
#pragma unroll
        for (int mt = 0; mt < 4; ++mt)
#pragma unroll
            for (int nt = 0; nt < 2; ++nt)
#pragma unroll
                for (int r = 0; r < 8; ++r)
                    C[(long)(mBase + mt * 16 + r) * ldc + nBase + nt * 16] =
                        (_Float16)acc[mt][nt][r];
    } else {
        float* C = (float*)Cp + (long)z * cBStride;
#pragma unroll
        for (int mt = 0; mt < 4; ++mt)
#pragma unroll
            for (int nt = 0; nt < 2; ++nt)
#pragma unroll
                for (int r = 0; r < 8; ++r)
                    C[(long)(mBase + mt * 16 + r) * ldc + nBase + nt * 16] =
                        acc[mt][nt][r];
    }
}

// ---------------- flash attention -----------------------------------------------
// grid: (NQ/64, HEADS, BATCH), 128 threads = 4 waves x 16 query rows.
// K tile row-major from kvbuf; V tile pre-transposed (Vt), both stage as b128.
__global__ __launch_bounds__(128) void attn_kernel(
    const _Float16* __restrict__ qh,   // [B][NQ][INNER], scale pre-folded
    const _Float16* __restrict__ kvh,  // [B][STOT][2*INNER] (k in cols 0..1023)
    const _Float16* __restrict__ vt,   // [B][H][DH][STOT]
    _Float16* __restrict__ oh) {       // [B][NQ][INNER]
    const int qt = blockIdx.x, h = blockIdx.y, b = blockIdx.z;
    __shared__ alignas(16) _Float16 sK[64][72];        // [key][dim]
    __shared__ alignas(16) _Float16 sV[64][72];        // [dim][key]
    __shared__ alignas(16) _Float16 sP[4][16][72];     // per-wave P staging

    const int tid  = threadIdx.x;
    const int lane = tid & 31;
    const int wave = tid >> 5;
    const int half = lane >> 4;
    const int l15  = lane & 15;

    // Q fragments (A layout), held for the whole kernel
    v16h aQ[2];
    {
        int qrow = qt * 64 + wave * 16 + l15;
        const _Float16* qp = qh + ((long)(b * NQ + qrow)) * INNER + h * DH;
#pragma unroll
        for (int kk = 0; kk < 2; ++kk)
            aQ[kk] = ld_frag(qp + kk * 32 + half * 8);
    }

    v8f accO[4];
#pragma unroll
    for (int dt = 0; dt < 4; ++dt) accO[dt] = {};
    float mrow[8], lrow[8];
#pragma unroll
    for (int r = 0; r < 8; ++r) { mrow[r] = -3.0e38f; lrow[r] = 0.f; }

    // tile-load indices: each thread owns 32 contiguous f16 of one row
    const int ldRow = tid >> 1;            // 0..63
    const int ldSeg = (tid & 1) * 32;      // 0 / 32

    const long kBase0 = ((long)b * STOT) * (2 * INNER) + h * DH;
    const long vBase0 = ((long)(b * HEADS + h) * DH) * STOT;

    for (int it = 0; it < STOT / 64; ++it) {
        __syncthreads();
        {
            const _Float16* kp =
                kvh + kBase0 + (long)(it * 64 + ldRow) * (2 * INNER) + ldSeg;
            const _Float16* vp = vt + vBase0 + (long)ldRow * STOT + it * 64 + ldSeg;
            _Float16* dk = &sK[ldRow][ldSeg];
            _Float16* dv = &sV[ldRow][ldSeg];
#pragma unroll
            for (int u = 0; u < 4; ++u) {
                *(h8*)(dk + u * 8) = *(const h8*)(kp + u * 8);
                *(h8*)(dv + u * 8) = *(const h8*)(vp + u * 8);
            }
        }
        __syncthreads();

        // S = Q * K^T  (4 key sub-tiles of 16)
        v8f sAcc[4];
#pragma unroll
        for (int jt = 0; jt < 4; ++jt) {
            v8f c = {};
#pragma unroll
            for (int kk = 0; kk < 2; ++kk) {
                int key = jt * 16 + l15;
                v16h bK = ld_frag(&sK[key][kk * 32 + half * 8]);
                c = wmma_f16(aQ[kk], bK, c);
            }
            sAcc[jt] = c;
        }

        // online softmax (rows m = r + 8*half; 16-lane halves share rows)
#pragma unroll
        for (int r = 0; r < 8; ++r) {
            float mx = sAcc[0][r];
#pragma unroll
            for (int jt = 1; jt < 4; ++jt) mx = fmaxf(mx, sAcc[jt][r]);
#pragma unroll
            for (int m = 8; m >= 1; m >>= 1) mx = fmaxf(mx, __shfl_xor(mx, m, 32));
            float mnew  = fmaxf(mrow[r], mx);
            float alpha = __expf(mrow[r] - mnew);
            float rsum = 0.f;
#pragma unroll
            for (int jt = 0; jt < 4; ++jt) {
                float p = __expf(sAcc[jt][r] - mnew);
                sAcc[jt][r] = p;
                rsum += p;
            }
#pragma unroll
            for (int m = 8; m >= 1; m >>= 1) rsum += __shfl_xor(rsum, m, 32);
            lrow[r] = lrow[r] * alpha + rsum;
            mrow[r] = mnew;
#pragma unroll
            for (int dt = 0; dt < 4; ++dt) accO[dt][r] *= alpha;
            // stage P (C layout -> LDS) for A-layout reload
            int mloc = r + half * 8;
#pragma unroll
            for (int jt = 0; jt < 4; ++jt)
                sP[wave][mloc][jt * 16 + l15] = (_Float16)sAcc[jt][r];
        }
        __syncthreads();

        // O += P * V
#pragma unroll
        for (int dt = 0; dt < 4; ++dt) {
#pragma unroll
            for (int kk = 0; kk < 2; ++kk) {
                v16h aP = ld_frag(&sP[wave][l15][kk * 32 + half * 8]);
                int dim = dt * 16 + l15;
                v16h bV = ld_frag(&sV[dim][kk * 32 + half * 8]);
                accO[dt] = wmma_f16(aP, bV, accO[dt]);
            }
        }
    }

    // normalize & store f16
#pragma unroll
    for (int dt = 0; dt < 4; ++dt) {
#pragma unroll
        for (int r = 0; r < 8; ++r) {
            int qrow = qt * 64 + wave * 16 + r + half * 8;
            float val = accO[dt][r] * (1.0f / lrow[r]);
            oh[((long)(b * NQ + qrow)) * INNER + h * DH + dt * 16 + l15] =
                (_Float16)val;
        }
    }
}

// ---------------- host orchestration -------------------------------------------
extern "C" void kernel_launch(void* const* d_in, const int* in_sizes, int n_in,
                              void* d_out, int out_size, void* d_ws, size_t ws_size,
                              hipStream_t stream) {
    (void)in_sizes; (void)n_in; (void)out_size; (void)ws_size;
    const float* x   = (const float*)d_in[0];
    const float* qs  = (const float*)d_in[1];
    const float* gx  = (const float*)d_in[2];
    const float* bx  = (const float*)d_in[3];
    const float* gq  = (const float*)d_in[4];
    const float* bq  = (const float*)d_in[5];
    const float* wq  = (const float*)d_in[6];
    const float* wkv = (const float*)d_in[7];
    const float* wo  = (const float*)d_in[8];
    float* out = (float*)d_out;

    char* ws = (char*)d_ws;
    size_t off = 0;
    auto take = [&](size_t bytes) {
        char* p = ws + off;
        off = (off + bytes + 255) & ~(size_t)255;
        return p;
    };
    _Float16* kvin  = (_Float16*)take((size_t)BATCH * STOT * DIMC * 2);
    _Float16* wqT   = (_Float16*)take((size_t)DIMC * INNER * 2);       // [N][K]
    _Float16* wkvT  = (_Float16*)take((size_t)DIMC * 2 * INNER * 2);   // [2N][K]
    _Float16* woT   = (_Float16*)take((size_t)INNER * DIMC * 2);       // [N][K]
    _Float16* qbuf  = (_Float16*)take((size_t)BATCH * NQ * INNER * 2);
    _Float16* kvbuf = (_Float16*)take((size_t)BATCH * STOT * 2 * INNER * 2);
    _Float16* vtbuf = (_Float16*)take((size_t)BATCH * HEADS * DH * STOT * 2);
    _Float16* att   = (_Float16*)take((size_t)BATCH * NQ * INNER * 2);

    const float scale = 0.125f;  // DH^-0.5, folded into w_q

    // 1) weights: f32 [K][N] -> f16 [N][K] (transposed)
    transpose_convert_kernel<<<dim3(DIMC / 64, INNER / 64), 256, 0, stream>>>(
        wq, wqT, DIMC, INNER, scale);
    transpose_convert_kernel<<<dim3(DIMC / 64, 2 * INNER / 64), 256, 0, stream>>>(
        wkv, wkvT, DIMC, 2 * INNER, 1.0f);
    transpose_convert_kernel<<<dim3(INNER / 64, DIMC / 64), 256, 0, stream>>>(
        wo, woT, INNER, DIMC, 1.0f);

    // 2) layernorm -> concatenated kv_input (xn rows then qn rows)
    layernorm_kernel<<<BATCH * STOT, 256, 0, stream>>>(x, qs, gx, bx, gq, bq, kvin);

    // 3) q = qn @ (w_q * scale)   (qn = tail NQ rows of kv_input)
    gemm_f16_kernel<true><<<dim3(INNER / 128, NQ / 128, BATCH), 256, 0, stream>>>(
        kvin + (long)TSEQ * DIMC, (long)STOT * DIMC, DIMC,
        wqT, DIMC,
        qbuf, INNER, (long)NQ * INNER, DIMC);

    // 4) kv = kv_input @ w_kv
    gemm_f16_kernel<true><<<dim3(2 * INNER / 128, STOT / 128, BATCH), 256, 0, stream>>>(
        kvin, (long)STOT * DIMC, DIMC,
        wkvT, DIMC,
        kvbuf, 2 * INNER, (long)STOT * 2 * INNER, DIMC);

    // 5) transpose V-half per head -> Vt[b][h][DH][STOT]
    transpose_v_kernel<<<dim3(STOT / 64, HEADS, BATCH), 256, 0, stream>>>(
        kvbuf, vtbuf);

    // 6) flash attention
    attn_kernel<<<dim3(NQ / 64, HEADS, BATCH), 128, 0, stream>>>(
        qbuf, kvbuf, vtbuf, att);

    // 7) final projection -> f32 output
    gemm_f16_kernel<false><<<dim3(DIMC / 128, NQ / 128, BATCH), 256, 0, stream>>>(
        att, (long)NQ * INNER, INNER,
        woT, DIMC,
        out, DIMC, (long)NQ * DIMC, INNER);
}
